// gen_color_stride_vgg16_83150566851429
// MI455X (gfx1250) — matrix operand order repaired and checked by reading, not compiled, not run on hardware
//
#include <hip/hip_runtime.h>
#include <hip/hip_bf16.h>

#define BATCH 8
#define CH    256
#define NREAL 2049
#define MREAL 2049
#define NPAD  2064   // 129 * 16
#define MPAD  2080   // 65 * 32
#define HWSZ  (112 * 112)
#define EPSV  1e-5f

typedef __attribute__((ext_vector_type(16))) __bf16 v16bf;
typedef __attribute__((ext_vector_type(8)))  __bf16 v8bf;
typedef __attribute__((ext_vector_type(8)))  float  v8f;

static __device__ __forceinline__ v16bf cat16(v8bf lo, v8bf hi) {
  return __builtin_shufflevector(lo, hi, 0, 1, 2, 3, 4, 5, 6, 7, 8, 9, 10, 11, 12, 13, 14, 15);
}

// ---------------------------------------------------------------- small prep
__global__ void init_counts_kernel(int* nt, int* nr) {
  if (threadIdx.x < 8) { nt[threadIdx.x] = 0; nr[threadIdx.x] = 0; }
}

__global__ void mask_max_kernel(const int* __restrict__ mask, int* __restrict__ cnt,
                                int perBatch, int total) {
  int i = blockIdx.x * blockDim.x + threadIdx.x;
  if (i >= total) return;
  atomicMax(&cnt[i / perBatch], mask[i] + 1);
}

// normalize over C (optional) + transpose (B,C,n) -> (B,nPad,C) + convert to bf16.
// Padded columns are written as zeros so downstream GEMMs never predicate loads.
__global__ void norm_convert_kernel(const float* __restrict__ x, __bf16* __restrict__ out,
                                    int colsReal, int colsPad, int doNorm) {
  int col = blockIdx.x * blockDim.x + threadIdx.x;
  if (col >= BATCH * colsPad) return;
  int b = col / colsPad, n = col % colsPad;
  __bf16* op = out + (size_t)col * CH;
  if (n >= colsReal) {
    for (int c = 0; c < CH; ++c) op[c] = (__bf16)0.f;
    return;
  }
  const float* p = x + (size_t)b * CH * colsReal + n;
  float mu = 0.f, rs = 1.f;
  if (doNorm) {
    float s = 0.f, ss = 0.f;
    for (int c = 0; c < CH; ++c) { float v = p[(size_t)c * colsReal]; s += v; ss += v * v; }
    mu = s * (1.0f / CH);
    rs = 1.0f / (sqrtf(fmaxf(ss - s * mu, 0.f)) + EPSV);
  }
  for (int c = 0; c < CH; ++c) op[c] = (__bf16)((p[(size_t)c * colsReal] - mu) * rs);
}

__global__ void wconv_kernel(const float* __restrict__ wq, const float* __restrict__ wk,
                             const float* __restrict__ wv, __bf16* oq, __bf16* ok, __bf16* ov) {
  int i = blockIdx.x * blockDim.x + threadIdx.x;
  if (i >= CH * CH) return;
  oq[i] = (__bf16)wq[i]; ok[i] = (__bf16)wk[i]; ov[i] = (__bf16)wv[i];
}

// ------------------------------------------------- q/k/v GEMM (WMMA bf16)
// out[n,o] = sum_c Xb[n,c] * W[o,c] + bias[o]; one wave per 16x16 tile.
// A fragment: two contiguous 8xbf16 chunks; B fragment: one contiguous 16xbf16 chunk.
__global__ __launch_bounds__(32) void qkv_gemm_kernel(
    const __bf16* __restrict__ Xb, const __bf16* __restrict__ W,
    const float* __restrict__ bias, __bf16* __restrict__ out, int rowsPad, int storeT) {
  int b = blockIdx.z, n0 = blockIdx.y * 16, o0 = blockIdx.x * 16;
  int lane = threadIdx.x & 31, r = lane & 15, h = lane >> 4;
  const __bf16* ap = Xb + ((size_t)b * rowsPad + n0 + r) * CH + h * 16;
  const __bf16* wp = W + (size_t)(o0 + r) * CH + h * 16;
  v8f acc = {};
#pragma unroll
  for (int ks = 0; ks < 8; ++ks) {            // K = 256 = 8 * 32
    v8bf  alo = *(const v8bf*)(ap + ks * 32);
    v8bf  ahi = *(const v8bf*)(ap + ks * 32 + 8);
    v16bf bf  = *(const v16bf*)(wp + ks * 32);
    acc = __builtin_amdgcn_wmma_f32_16x16x32_bf16(false, cat16(alo, ahi), false, bf,
                                                  (short)0, acc, false, false);
  }
  float bb = bias[o0 + r];
  if (storeT) {                                // [b][o][m] layout (for v)
    __bf16* op = out + ((size_t)b * CH + o0 + r) * rowsPad + n0 + 8 * h;
#pragma unroll
    for (int rr = 0; rr < 8; ++rr) op[rr] = (__bf16)(acc[rr] + bb);
  } else {                                     // [b][n][o] layout (for q,k)
    __bf16* op = out + ((size_t)b * rowsPad + n0 + 8 * h) * CH + o0 + r;
#pragma unroll
    for (int rr = 0; rr < 8; ++rr) op[(size_t)rr * CH] = (__bf16)(acc[rr] + bb);
  }
}

// --------------------------- fused sim -> softmax -> attn out -> wg (WMMA)
__global__ __launch_bounds__(256) void attn_wg_kernel(
    const __bf16* __restrict__ qbf, const __bf16* __restrict__ kbf, const __bf16* __restrict__ vbf,
    const int* __restrict__ ntp, const int* __restrict__ nrp,
    float* __restrict__ outWg, float* __restrict__ outAttn) {
  extern __shared__ char smem[];
  float*  sim    = (float*)smem;                              // 16 x MPAD f32
  __bf16* attnbf = (__bf16*)(smem + 16 * MPAD * 4);           // 16 x MPAD bf16
  float*  red    = (float*)(smem + 16 * MPAD * 6);            // 16 x 16 f32
  int b = blockIdx.y, n0 = blockIdx.x * 16;
  int tid = threadIdx.x, wave = tid >> 5, lane = tid & 31;
  int r = lane & 15, h = lane >> 4;
  int nrb = nrp[b], ntb = ntp[b];

  // preload q A-fragments for the 16-row stripe (K = 256 -> 8 fragments)
  v16bf afrag[8];
  {
    const __bf16* qp = qbf + ((size_t)b * NPAD + (n0 + r)) * CH + h * 16;
#pragma unroll
    for (int ks = 0; ks < 8; ++ks) {
      v8bf lo = *(const v8bf*)(qp + ks * 32);
      v8bf hi = *(const v8bf*)(qp + ks * 32 + 8);
      afrag[ks] = cat16(lo, hi);
    }
  }

  // ---- sim tiles across waves ----
  for (int t = wave; t < MPAD / 16; t += 8) {
    int m0 = t * 16;
    const __bf16* kp = kbf + ((size_t)b * MPAD + (m0 + r)) * CH + h * 16;
    v8f acc = {};
#pragma unroll
    for (int ks = 0; ks < 8; ++ks) {
      v16bf bf = *(const v16bf*)(kp + ks * 32);
      acc = __builtin_amdgcn_wmma_f32_16x16x32_bf16(false, afrag[ks], false, bf,
                                                    (short)0, acc, false, false);
    }
    int col = m0 + r;
    bool colOK = (col < nrb);
#pragma unroll
    for (int rr = 0; rr < 8; ++rr) {
      float s = acc[rr] * 100.0f;              // / TEMP
      if (!colOK) s = -3.0e38f;
      sim[(rr + 8 * h) * MPAD + col] = s;
    }
  }
  __syncthreads();

  // ---- row softmax: 16 threads per row ----
  int row = tid >> 4, sub = tid & 15;
  float mx = -3.0e38f;
  for (int m = sub; m < MPAD; m += 16) mx = fmaxf(mx, sim[row * MPAD + m]);
  red[row * 16 + sub] = mx;
  __syncthreads();
  float rm = -3.0e38f;
#pragma unroll
  for (int i = 0; i < 16; ++i) rm = fmaxf(rm, red[row * 16 + i]);
  __syncthreads();
  float sum = 0.f;
  for (int m = sub; m < MPAD; m += 16) {
    float e = __expf(sim[row * MPAD + m] - rm);
    sim[row * MPAD + m] = e;
    sum += e;
  }
  red[row * 16 + sub] = sum;
  __syncthreads();
  float rsum = 0.f;
#pragma unroll
  for (int i = 0; i < 16; ++i) rsum += red[row * 16 + i];
  int nRow = n0 + row;
  float inv = ((nRow < ntb) ? 1.0f : 0.0f) / rsum;
  for (int m = sub; m < MPAD; m += 16) {
    float a = sim[row * MPAD + m] * inv;
    sim[row * MPAD + m] = a;
    attnbf[row * MPAD + m] = (__bf16)a;        // bf16 stripe for the wg GEMM
  }
  if (nRow < NREAL) {
    float* arow = outAttn + ((size_t)b * NREAL + nRow) * MREAL;
    for (int m = sub; m < MREAL; m += 16) arow[m] = sim[row * MPAD + m];
  }
  __syncthreads();

  // ---- wg = attn (16 x MPAD, bf16 LDS) @ v (MPAD x 256); 2 column-tiles/wave ----
#pragma unroll
  for (int ci = 0; ci < 2; ++ci) {
    int o0 = (wave + ci * 8) * 16;
    const __bf16* vp  = vbf + ((size_t)b * CH + (o0 + r)) * MPAD + h * 16;  // [b][c][m]
    const __bf16* abp = attnbf + r * MPAD + h * 16;
    v8f acc = {};
#pragma unroll 5
    for (int ks = 0; ks < MPAD / 32; ++ks) {   // 65 k-steps
      v8bf  lo = *(const v8bf*)(abp + ks * 32);
      v8bf  hi = *(const v8bf*)(abp + ks * 32 + 8);
      v16bf bf = *(const v16bf*)(vp + ks * 32);
      acc = __builtin_amdgcn_wmma_f32_16x16x32_bf16(false, cat16(lo, hi), false, bf,
                                                    (short)0, acc, false, false);
    }
    float* op = outWg + ((size_t)b * CH + o0 + r) * NREAL + n0 + 8 * h;
#pragma unroll
    for (int rr = 0; rr < 8; ++rr)
      if (n0 + 8 * h + rr < NREAL) op[rr] = acc[rr];   // transposed (B,C,N) out
  }
}

__global__ void zero_alpha_kernel(float* p, int n) {
  int i = blockIdx.x * blockDim.x + threadIdx.x;
  if (i < n) p[i] = 0.f;
}

// ------------------------------------------------------------------- launch
extern "C" void kernel_launch(void* const* d_in, const int* in_sizes, int n_in,
                              void* d_out, int out_size, void* d_ws, size_t ws_size,
                              hipStream_t stream) {
  const float* tgt  = (const float*)d_in[0];
  const float* refc = (const float*)d_in[1];
  const int*   mskT = (const int*)d_in[2];
  const int*   mskR = (const int*)d_in[3];
  const float* rgb  = (const float*)d_in[5];
  const float* Wq = (const float*)d_in[6];  const float* bq = (const float*)d_in[7];
  const float* Wk = (const float*)d_in[8];  const float* bk = (const float*)d_in[9];
  const float* Wv = (const float*)d_in[10]; const float* bv = (const float*)d_in[11];
  (void)in_sizes; (void)n_in; (void)out_size; (void)ws_size;

  char* ws = (char*)d_ws;
  size_t off = 0;
  auto alloc = [&](size_t bytes) { size_t o = off; off += (bytes + 255) & ~(size_t)255; return o; };
  int*    nT  = (int*)(ws + alloc(8 * 4));
  int*    nR  = (int*)(ws + alloc(8 * 4));
  __bf16* WqB = (__bf16*)(ws + alloc((size_t)CH * CH * 2));
  __bf16* WkB = (__bf16*)(ws + alloc((size_t)CH * CH * 2));
  __bf16* WvB = (__bf16*)(ws + alloc((size_t)CH * CH * 2));
  __bf16* nTb = (__bf16*)(ws + alloc((size_t)BATCH * NPAD * CH * 2));  // norm(target) [b][n][c]
  __bf16* nRb = (__bf16*)(ws + alloc((size_t)BATCH * MPAD * CH * 2));  // norm(ref)    [b][m][c]
  __bf16* gRb = (__bf16*)(ws + alloc((size_t)BATCH * MPAD * CH * 2));  // rgb_ref      [b][m][c]
  __bf16* qB  = (__bf16*)(ws + alloc((size_t)BATCH * NPAD * CH * 2));  // q [b][n][c]
  __bf16* kB  = (__bf16*)(ws + alloc((size_t)BATCH * MPAD * CH * 2));  // k [b][m][c]
  __bf16* vB  = (__bf16*)(ws + alloc((size_t)BATCH * MPAD * CH * 2));  // v [b][c][m]

  init_counts_kernel<<<1, 32, 0, stream>>>(nT, nR);
  int totalMask = BATCH * HWSZ;
  mask_max_kernel<<<(totalMask + 255) / 256, 256, 0, stream>>>(mskT, nT, HWSZ, totalMask);
  mask_max_kernel<<<(totalMask + 255) / 256, 256, 0, stream>>>(mskR, nR, HWSZ, totalMask);
  norm_convert_kernel<<<(BATCH * NPAD + 127) / 128, 128, 0, stream>>>(tgt,  nTb, NREAL, NPAD, 1);
  norm_convert_kernel<<<(BATCH * MPAD + 127) / 128, 128, 0, stream>>>(refc, nRb, MREAL, MPAD, 1);
  norm_convert_kernel<<<(BATCH * MPAD + 127) / 128, 128, 0, stream>>>(rgb,  gRb, MREAL, MPAD, 0);
  wconv_kernel<<<(CH * CH + 255) / 256, 256, 0, stream>>>(Wq, Wk, Wv, WqB, WkB, WvB);

  dim3 gq(CH / 16, NPAD / 16, BATCH);
  dim3 gk(CH / 16, MPAD / 16, BATCH);
  qkv_gemm_kernel<<<gq, 32, 0, stream>>>(nTb, WqB, bq, qB, NPAD, 0);
  qkv_gemm_kernel<<<gk, 32, 0, stream>>>(nRb, WkB, bk, kB, MPAD, 0);
  qkv_gemm_kernel<<<gk, 32, 0, stream>>>(gRb, WvB, bv, vB, MPAD, 1);

  float* outWg    = (float*)d_out;
  float* outAttn  = outWg   + (size_t)BATCH * CH * NREAL;
  float* outAlpha = outAttn + (size_t)BATCH * NREAL * MREAL;
  size_t ldsBytes = (size_t)16 * MPAD * 6 + 16 * 16 * 4;   // ~196 KB of 320 KB/WGP
  dim3 ga(NPAD / 16, BATCH);
  attn_wg_kernel<<<ga, 256, ldsBytes, stream>>>(qB, kB, vB, nT, nR, outWg, outAttn);
  zero_alpha_kernel<<<(BATCH * NREAL + 255) / 256, 256, 0, stream>>>(outAlpha, BATCH * NREAL);
}
